// GroupingExclusiveMLP1_29884382446131
// MI455X (gfx1250) — matrix operand order repaired and checked by reading, not compiled, not run on hardware
//
#include <hip/hip_runtime.h>
#include <hip/hip_bf16.h>

#define BASIC   13
#define NGRP    21
#define GS      8
#define NCOL    (BASIC + NGRP*GS)   // 181
#define HDIM    512
#define ROWS    128                 // rows per workgroup
#define THREADS 256
#define MSTR    40                  // mixed LDS row stride (f16): 80B, 16B-aligned, conflict-free
#define WSTR    40                  // w1h LDS row stride (f16)

typedef __attribute__((ext_vector_type(16))) _Float16 v16h;
typedef __attribute__((ext_vector_type(8)))  _Float16 v8h;
typedef __attribute__((ext_vector_type(8)))  float    v8f;
typedef __attribute__((ext_vector_type(4)))  float    v4f;

__device__ __forceinline__ float sigmoidf(float x) {
    return 1.0f / (1.0f + __expf(-x));
}

__global__ __launch_bounds__(THREADS) void fused_group_mlp(
    const float* __restrict__ data, const float* __restrict__ gW,
    const float* __restrict__ gb,   const float* __restrict__ W1,
    const float* __restrict__ b1,   const float* __restrict__ W2,
    const float* __restrict__ b2,   float* __restrict__ out)
{
    __shared__ __align__(16) _Float16 mixedh[ROWS * MSTR];   // 10240 B: [row][k], k=0..33 used
    __shared__ __align__(16) _Float16 w1h[HDIM * WSTR];      // 40960 B: [n][k], k=0..31 used
    __shared__ __align__(16) v4f      coefs[HDIM];           //  8192 B: {b1, W2, W1[32], W1[33]}[n]
    // total LDS = 59392 B

    const int tid  = threadIdx.x;
    const int row0 = blockIdx.x * ROWS;

    // ---------------- phase 0: weights -> LDS (L2-resident, coalesced) ----------------
    for (int idx = tid; idx < 32 * HDIM; idx += THREADS) {
        const int k = idx >> 9;            // 0..31
        const int n = idx & (HDIM - 1);    // 0..511 (fastest -> coalesced global read)
        w1h[n * WSTR + k] = (_Float16)W1[k * HDIM + n];
    }
    for (int n = tid; n < HDIM; n += THREADS) {
        v4f c;
        c.x = b1[n];
        c.y = W2[n];
        c.z = W1[32 * HDIM + n];
        c.w = W1[33 * HDIM + n];
        coefs[n] = c;
    }

    // ---------------- phase 1: group Linear+sigmoid -> mixed f16 tile ----------------
    {
        const int r = tid >> 1;            // row within tile (2 threads / row)
        const int h = tid & 1;
        const float* __restrict__ drow = data + (size_t)(row0 + r) * NCOL;
        _Float16* mrow = mixedh + r * MSTR;
        if (h == 0) {
            #pragma unroll
            for (int c = 0; c < BASIC; ++c)
                mrow[c] = (_Float16)__builtin_nontemporal_load(drow + c);
            #pragma unroll
            for (int g = 0; g < 10; ++g) {
                float a = gb[g];
                const float* dp = drow + BASIC + g * GS;
                const float* wp = gW + g * GS;
                #pragma unroll
                for (int d = 0; d < GS; ++d)
                    a = fmaf(__builtin_nontemporal_load(dp + d), wp[d], a);
                mrow[BASIC + g] = (_Float16)sigmoidf(a);
            }
        } else {
            #pragma unroll
            for (int g = 10; g < NGRP; ++g) {
                float a = gb[g];
                const float* dp = drow + BASIC + g * GS;
                const float* wp = gW + g * GS;
                #pragma unroll
                for (int d = 0; d < GS; ++d)
                    a = fmaf(__builtin_nontemporal_load(dp + d), wp[d], a);
                mrow[BASIC + g] = (_Float16)sigmoidf(a);
            }
        }
    }

    __syncthreads();

    // ---------------- phase 2: WMMA layer1 (K=32) + VALU K-tail, fused layer2 ----------------
    const int lane = tid & 31;
    const int wv   = tid >> 5;          // 8 waves, 16 rows each
    const int m0   = wv * 16;
    const int ml   = lane & 15;         // N index within tile / M index for A rows
    const int kh   = lane >> 4;         // K-half select (ISA §7.12.2 layouts)

    // A fragment, K = 0..31 (loop invariant).
    // Lane layout: elems 0..7 = K (8*kh .. 8*kh+7), elems 8..15 = K (16+8*kh ..).
    const _Float16* mrow = mixedh + (m0 + ml) * MSTR;
    v16h a0;
    {
        v8h lo = *(const v8h*)(mrow + kh * 8);
        v8h hi = *(const v8h*)(mrow + 16 + kh * 8);
        #pragma unroll
        for (int i = 0; i < 8; ++i) { a0[i] = lo[i]; a0[8 + i] = hi[i]; }
    }

    // K-tail features 32,33 for the 8 rows this lane's accumulators cover
    // (C layout: VGPR v <-> M = v + 8*kh, N = lane%16). Loop invariant -> hoist.
    float m32f[8], m33f[8];
    #pragma unroll
    for (int v = 0; v < 8; ++v) {
        const _Float16* p = mixedh + (m0 + v + 8 * kh) * MSTR + 32;
        m32f[v] = (float)p[0];
        m33f[v] = (float)p[1];
    }

    float acc[8];
    #pragma unroll
    for (int v = 0; v < 8; ++v) acc[v] = 0.0f;

    const int bh = kh * 8;              // B K-half: elems 0..7 = K (bh .. bh+7), elems 8..15 = K (16+bh ..)
    for (int t = 0; t < HDIM / 16; ++t) {
        const int n = t * 16 + ml;
        const _Float16* wr = w1h + n * WSTR + bh;
        v16h b0;
        {
            v8h lo = *(const v8h*)(wr);
            v8h hi = *(const v8h*)(wr + 16);
            #pragma unroll
            for (int i = 0; i < 8; ++i) { b0[i] = lo[i]; b0[8 + i] = hi[i]; }
        }
        const v4f cf = coefs[n];        // {b1[n], W2[n], W1[32][n], W1[33][n]} - one ds_load_b128

        v8f c = {};
        c = __builtin_amdgcn_wmma_f32_16x16x32_f16(false, a0, false, b0, (short)0, c, false, false);

        // h_pre = c + b1 + m32*W1[32][n] + m33*W1[33][n]; acc += relu(h_pre) * W2[n]
        #pragma unroll
        for (int v = 0; v < 8; ++v) {
            float hp = c[v] + cf.x;
            hp = fmaf(m32f[v], cf.z, hp);
            hp = fmaf(m33f[v], cf.w, hp);
            acc[v] = fmaf(fmaxf(hp, 0.0f), cf.y, acc[v]);
        }
    }

    // Reduce across the 16 lanes of each half (lanes 0-15 hold M=v, 16-31 hold M=v+8).
    #pragma unroll
    for (int off = 8; off >= 1; off >>= 1) {
        #pragma unroll
        for (int v = 0; v < 8; ++v)
            acc[v] += __shfl_xor(acc[v], off, 32);
    }

    if (ml == 0) {
        const float b2v = b2[0];
        #pragma unroll
        for (int v = 0; v < 8; ++v) {
            const float y = sigmoidf(acc[v] + b2v);
            __builtin_nontemporal_store(y, out + row0 + m0 + kh * 8 + v);
        }
    }
}

extern "C" void kernel_launch(void* const* d_in, const int* in_sizes, int n_in,
                              void* d_out, int out_size, void* d_ws, size_t ws_size,
                              hipStream_t stream) {
    (void)n_in; (void)out_size; (void)d_ws; (void)ws_size;
    const float* data = (const float*)d_in[0];
    const float* gW   = (const float*)d_in[1];
    const float* gb   = (const float*)d_in[2];
    const float* W1   = (const float*)d_in[3];
    const float* b1   = (const float*)d_in[4];
    const float* W2   = (const float*)d_in[5];
    const float* b2   = (const float*)d_in[6];
    float* out        = (float*)d_out;

    const int B = in_sizes[0] / NCOL;          // 262144
    dim3 grid(B / ROWS);                       // 2048 workgroups
    fused_group_mlp<<<grid, THREADS, 0, stream>>>(data, gW, gb, W1, b1, W2, b2, out);
}
